// CustomLSTM_49314814493039
// MI455X (gfx1250) — compile-verified, matrix operand order checked by reading, not run on hardware
//
#include <hip/hip_runtime.h>
#include <hip/hip_bf16.h>
#include <cstdint>
#include <cstddef>

typedef __attribute__((ext_vector_type(16))) _Float16 v16h;
typedef __attribute__((ext_vector_type(8)))  _Float16 v8h;
typedef __attribute__((ext_vector_type(8)))  float    v8f;

namespace {
constexpr int T_  = 512;
constexpr int B_  = 64;
constexpr int IN_ = 512;
constexpr int H_  = 512;
constexpr int G4  = 4 * H_;      // 2048 gate columns
constexpr int KC  = IN_ + H_;    // 1024 fused K ([x|h] or [h_lower|h_prev])
}

union Frag { v16h v; v8h h[2]; };

__device__ __forceinline__ float sigf(float x) { return 1.0f / (1.0f + __expf(-x)); }

// ---------------- prep kernels ----------------

__global__ void cvt_f16_kernel(const float* __restrict__ src, _Float16* __restrict__ dst, int n) {
  int i = blockIdx.x * blockDim.x + threadIdx.x;
  int stride = gridDim.x * blockDim.x;
  for (; i < n; i += stride) dst[i] = (_Float16)src[i];
}

// Wcat[n][k] = k<512 ? Wi[n][k] : Wh[n][k-512]   (2048 x 1024, f16)
__global__ void pack_wcat_kernel(const float* __restrict__ Wi, const float* __restrict__ Wh,
                                 _Float16* __restrict__ Wcat) {
  int idx = blockIdx.x * blockDim.x + threadIdx.x;   // over 2048*1024
  int n = idx >> 10, k = idx & 1023;
  float v = (k < IN_) ? Wi[n * IN_ + k] : Wh[n * H_ + (k - IN_)];
  Wcat[idx] = (_Float16)v;
}

__global__ void bias_sum_kernel(const float* __restrict__ a, const float* __restrict__ b,
                                float* __restrict__ o) {
  int i = blockIdx.x * blockDim.x + threadIdx.x;
  if (i < G4) o[i] = a[i] + b[i];
}

__global__ void zero_kernel(uint32_t* __restrict__ p, int n) {
  int i = blockIdx.x * blockDim.x + threadIdx.x;
  if (i < n) p[i] = 0u;
}

// ---------------- per-timestep fused LSTM layer ----------------
// gates(64x2048) = [Alo | Ahi](64x1024) @ Wcat^T(1024x2048) + bsum, then LSTM cell.
// Grid: 16 blocks x 256 threads = 128 waves. Wave tile: 16 rows x 16 units, all 4 gates.
__global__ __launch_bounds__(256) void lstm_step_kernel(
    const _Float16* __restrict__ Alo,    // (64 x 512) x_t (or h_lower)
    const _Float16* __restrict__ Ahi,    // (64 x 512) h_{t-1} (f16)
    const _Float16* __restrict__ Wcat,   // (2048 x 1024) [Wi|Wh] row-major f16
    const float*    __restrict__ bsum,   // (2048) bi+bh
    float*          __restrict__ c_state,// (64 x 512) in/out
    float*          __restrict__ hf32,   // (64 x 512) h out (f32)
    _Float16*       __restrict__ hf16)   // (64 x 512) h out (f16, next step's A)
{
  const int wave  = blockIdx.x * 8 + (threadIdx.x >> 5);
  const int lane  = threadIdx.x & 31;
  const int mtile = wave & 3;    // 4 tiles of 16 rows -> M=64
  const int utile = wave >> 2;   // 32 tiles of 16 units -> H=512
  const int lrow  = lane & 15;
  const int lhalf = lane >> 4;

  const int ucol = utile * 16 + lrow;   // unit column (also C/D column = lane&15)
  const int arow = mtile * 16 + lrow;   // A row

  // Accumulators, one per gate, initialized with the (bi+bh) bias.
  v8f acc[4];
#pragma unroll
  for (int g = 0; g < 4; ++g) {
    float bv = bsum[g * H_ + ucol];
#pragma unroll
    for (int j = 0; j < 8; ++j) acc[g][j] = bv;
  }

  const _Float16* aLo = Alo + (size_t)arow * 512;
  const _Float16* aHi = Ahi + (size_t)arow * 512;
  const _Float16* brow[4];
#pragma unroll
  for (int g = 0; g < 4; ++g) {
    brow[g] = Wcat + (size_t)(g * H_ + ucol) * KC;   // weight row for this output col
    __builtin_prefetch(brow[g], 0, 3);               // global_prefetch_b8
  }

  // K loop: 32 chunks of 32. A/B fragment layout per ISA 16-bit tables:
  //   elems 0..7  = K (lhalf?8:0)..+7 ; elems 8..15 = K 16+(lhalf?8:0)..+7
  for (int kt = 0; kt < KC / 32; ++kt) {
    const int k0 = kt * 32;
    const _Float16* ab = (k0 < IN_) ? (aLo + k0) : (aHi + (k0 - IN_));
    Frag a;
    a.h[0] = *(const v8h*)(ab + lhalf * 8);
    a.h[1] = *(const v8h*)(ab + 16 + lhalf * 8);
    Frag b[4];
#pragma unroll
    for (int g = 0; g < 4; ++g) {
      const _Float16* bb = brow[g] + k0;
      b[g].h[0] = *(const v8h*)(bb + lhalf * 8);
      b[g].h[1] = *(const v8h*)(bb + 16 + lhalf * 8);
    }
#pragma unroll
    for (int g = 0; g < 4; ++g)
      acc[g] = __builtin_amdgcn_wmma_f32_16x16x32_f16(
          /*neg_a=*/false, a.v, /*neg_b=*/false, b[g].v,
          /*c_mod=*/(short)0, acc[g], /*reuse_a=*/false, /*reuse_b=*/false);
  }

  // Elementwise LSTM cell. C/D layout: VGPR j, lane l -> row = j + (l>=16)*8, col = l&15.
  const int mbase = mtile * 16 + lhalf * 8;
#pragma unroll
  for (int j = 0; j < 8; ++j) {
    const int m   = mbase + j;
    const int off = m * H_ + ucol;
    float iv = sigf(acc[0][j]);
    float fv = sigf(acc[1][j]);
    float ov = sigf(acc[2][j]);
    float gv = tanhf(acc[3][j]);
    float cn = fv * c_state[off] + iv * gv;
    float hv = ov * tanhf(cn);
    c_state[off] = cn;
    hf32[off]    = hv;
    hf16[off]    = (_Float16)hv;
  }
}

// Pack hT (L,B,H) and cT (L,B,H) after the outputs block.
__global__ void finalize_kernel(const float* __restrict__ h0, const float* __restrict__ c0,
                                const float* __restrict__ c1, float* __restrict__ out) {
  const int BH = B_ * H_;
  int i = blockIdx.x * blockDim.x + threadIdx.x;
  if (i >= 2 * BH) return;
  const size_t OUT0 = (size_t)T_ * BH;
  float hv = (i < BH) ? h0[i] : out[(size_t)(T_ - 1) * BH + (i - BH)];  // hT1 == outputs[T-1]
  out[OUT0 + i] = hv;
  float cv = (i < BH) ? c0[i] : c1[i - BH];
  out[OUT0 + 2 * BH + i] = cv;
}

// ---------------- host ----------------

extern "C" void kernel_launch(void* const* d_in, const int* in_sizes, int n_in,
                              void* d_out, int out_size, void* d_ws, size_t ws_size,
                              hipStream_t stream) {
  (void)in_sizes; (void)n_in; (void)out_size; (void)ws_size;
  const float* x   = (const float*)d_in[0];
  const float* Wi0 = (const float*)d_in[1];
  const float* bi0 = (const float*)d_in[2];
  const float* Wh0 = (const float*)d_in[3];
  const float* bh0 = (const float*)d_in[4];
  const float* Wi1 = (const float*)d_in[5];
  const float* bi1 = (const float*)d_in[6];
  const float* Wh1 = (const float*)d_in[7];
  const float* bh1 = (const float*)d_in[8];
  float* out = (float*)d_out;

  const int BH = B_ * H_;
  char* w = (char*)d_ws;
  _Float16* x16   = (_Float16*)w;  w += (size_t)T_ * B_ * IN_ * 2;   // 32 MB
  _Float16* W0cat = (_Float16*)w;  w += (size_t)G4 * KC * 2;         // 4 MB
  _Float16* W1cat = (_Float16*)w;  w += (size_t)G4 * KC * 2;         // 4 MB
  float* bsum0    = (float*)w;     w += (size_t)G4 * 4;
  float* bsum1    = (float*)w;     w += (size_t)G4 * 4;
  float* c0       = (float*)w;     w += (size_t)BH * 4;   // zeroed block starts here
  float* c1       = (float*)w;     w += (size_t)BH * 4;
  float* h0f32    = (float*)w;     w += (size_t)BH * 4;
  _Float16* h0f16[2];
  h0f16[0] = (_Float16*)w;         w += (size_t)BH * 2;
  h0f16[1] = (_Float16*)w;         w += (size_t)BH * 2;
  _Float16* h1f16[2];
  h1f16[0] = (_Float16*)w;         w += (size_t)BH * 2;
  h1f16[1] = (_Float16*)w;         w += (size_t)BH * 2;

  // Prep: f16 conversions, weight packing, bias sums, zero states.
  cvt_f16_kernel<<<4096, 256, 0, stream>>>(x, x16, T_ * B_ * IN_);
  pack_wcat_kernel<<<(G4 * KC) / 256, 256, 0, stream>>>(Wi0, Wh0, W0cat);
  pack_wcat_kernel<<<(G4 * KC) / 256, 256, 0, stream>>>(Wi1, Wh1, W1cat);
  bias_sum_kernel<<<8, 256, 0, stream>>>(bi0, bh0, bsum0);
  bias_sum_kernel<<<8, 256, 0, stream>>>(bi1, bh1, bsum1);
  const int zwords = (3 * BH * 4 + 4 * BH * 2) / 4;  // c0,c1,h0f32,h0f16[2],h1f16[2]
  zero_kernel<<<(zwords + 255) / 256, 256, 0, stream>>>((uint32_t*)c0, zwords);

  // Sequential recurrence: 2 fused-GEMM kernels per timestep, h f16 ping-ponged.
  for (int t = 0; t < T_; ++t) {
    const int p = t & 1;
    // Layer 0: A = [x_t | h0_{t-1}]
    lstm_step_kernel<<<16, 256, 0, stream>>>(
        x16 + (size_t)t * B_ * IN_, h0f16[p], W0cat, bsum0,
        c0, h0f32, h0f16[1 - p]);
    // Layer 1: A = [h0_t | h1_{t-1}] ; h f32 goes straight to outputs[t]
    lstm_step_kernel<<<16, 256, 0, stream>>>(
        h0f16[1 - p], h1f16[p], W1cat, bsum1,
        c1, out + (size_t)t * BH, h1f16[1 - p]);
  }

  finalize_kernel<<<(2 * BH + 255) / 256, 256, 0, stream>>>(h0f32, c0, c1, out);
}
